// BooleanReservoir_31284541784138
// MI455X (gfx1250) — matrix operand order repaired and checked by reading, not compiled, not run on hardware
//
#include <hip/hip_runtime.h>
#include <stdint.h>
#include <math.h>

typedef __attribute__((ext_vector_type(16))) _Float16 v16h;
typedef __attribute__((ext_vector_type(8)))  float    v8f;

#define I_N   512
#define R_N   16384
#define NN    16896     // I_N + R_N
#define KADJ  8
#define MB    64
#define SS    16
#define NOUT  10

// ---- workspace byte offsets (all 32B aligned where needed) ----
// wbits   : 1024  u32  @ 0        (4 KB)
// xbits   : 2048  u32  @ 4096     (8 KB)
// adj16   : NN*8  u16  @ 12288    (270336 B)
// lutbits : NN*8  u32  @ 282624   (540672 B)
// Wh      : 16*R_N f16 @ 823296   (524288 B)   zero-padded cols 10..15
// sres_sw : MB*R_N f16 @ 1347584  (2 MB)       WMMA-A-swizzled reservoir states
#define OFF_WBITS   0
#define OFF_XBITS   4096
#define OFF_ADJ16   12288
#define OFF_LUTB    282624
#define OFF_WH      823296
#define OFF_SRES    1347584

// ------------------------------------------------------------------
// Prep: bit-pack w_in columns and x, mask+pack adjacency to u16,
// bit-pack LUT, convert W_out to f16 (padded to 16 rows).
// ------------------------------------------------------------------
__global__ __launch_bounds__(256)
void prep_kernel(const float* __restrict__ x, const float* __restrict__ w_in,
                 const int* __restrict__ adj_list, const int* __restrict__ deg,
                 const float* __restrict__ lut, const float* __restrict__ W_out,
                 uint32_t* __restrict__ wbits, uint32_t* __restrict__ xbits,
                 uint16_t* __restrict__ adj16, uint32_t* __restrict__ lutbits,
                 _Float16* __restrict__ Wh) {
  int t = blockIdx.x * blockDim.x + threadIdx.x;
  if (t < 1024) {                       // wbits[c*512+j]
    int c = t >> 9, j = t & 511;
    uint32_t v = 0;
    #pragma unroll 4
    for (int k = 0; k < 32; ++k)
      if (w_in[(c * 32 + k) * 512 + j] != 0.f) v |= (1u << k);
    wbits[t] = v;
    return;
  }
  t -= 1024;
  if (t < 2048) {                       // xbits[(m*16+s)*2+c]
    uint32_t v = 0;
    const float* xp = x + t * 32;
    #pragma unroll 4
    for (int b = 0; b < 32; ++b)
      if (xp[b] != 0.f) v |= (1u << b);
    xbits[t] = v;
    return;
  }
  t -= 2048;
  if (t < NN * KADJ) {                  // adj16: masked neighbors -> dummy node NN (state 0)
    int n = t >> 3, k = t & 7;
    int a = (k < deg[n]) ? adj_list[t] : NN;
    adj16[t] = (uint16_t)a;
    return;
  }
  t -= NN * KADJ;
  if (t < NN * 8) {                     // lutbits word: bit i = lut[n*256 + w*32 + i]
    const float* lp = lut + t * 32;
    uint32_t v = 0;
    #pragma unroll 4
    for (int i = 0; i < 32; ++i)
      if (lp[i] != 0.f) v |= (1u << i);
    lutbits[t] = v;
    return;
  }
  t -= NN * 8;
  if (t < 16 * R_N) {                   // Wh[col][k], cols 10..15 zero
    int col = t >> 14, k = t & (R_N - 1);
    float f = (col < NOUT) ? W_out[col * R_N + k] : 0.f;
    Wh[t] = (_Float16)f;
    return;
  }
}

// ------------------------------------------------------------------
// Reservoir: one workgroup per batch element. Boolean state double-
// buffered in LDS; 64 synchronous ticks with workgroup barriers.
// Final states written pre-swizzled for the WMMA A-fragment layout.
// ------------------------------------------------------------------
__global__ __launch_bounds__(1024)
void reservoir_kernel(const float* __restrict__ initial_states,
                      const uint32_t* __restrict__ wbits,
                      const uint32_t* __restrict__ xbits,
                      const uint16_t* __restrict__ adj16,
                      const uint32_t* __restrict__ lutbits,
                      uint16_t* __restrict__ sres_sw) {
  __shared__ uint8_t sbuf[2][16928];    // NN + dummy, padded
  const int m   = blockIdx.x;
  const int tid = threadIdx.x;

  for (int n = tid; n < NN; n += 1024)
    sbuf[0][n] = (initial_states[n] != 0.f) ? (uint8_t)1 : (uint8_t)0;
  if (tid == 0) { sbuf[0][NN] = 0; sbuf[1][NN] = 0; }   // dummy node always 0
  __syncthreads();

  int cur = 0;
  for (int s = 0; s < SS; ++s) {
    for (int c = 0; c < 2; ++c) {
      // injection: states[:512] ^= parity(x_bits & w_col_bits)
      if (tid < I_N) {
        uint32_t xb = xbits[(m * SS + s) * 2 + c];
        uint32_t wb = wbits[c * I_N + tid];
        sbuf[cur][tid] ^= (uint8_t)(__popc(xb & wb) & 1);
      }
      __syncthreads();
      for (int t2 = 0; t2 < 2; ++t2) {
        const uint8_t* sc = sbuf[cur];
        uint8_t*       sn = sbuf[cur ^ 1];
        #pragma unroll 2
        for (int n = tid; n < NN; n += 1024) {
          // prefetch next iteration's adjacency + LUT rows (L2 -> near cache)
          if (n + 1024 < NN) {
            __builtin_prefetch(adj16 + (n + 1024) * 8, 0, 3);
            __builtin_prefetch(lutbits + (n + 1024) * 8, 0, 3);
          }
          uint4 aw = *(const uint4*)(adj16 + n * 8);     // 8 x u16 neighbor ids
          uint32_t idx =
              ((uint32_t)sc[aw.x & 0xffffu] << 7) | ((uint32_t)sc[aw.x >> 16] << 6) |
              ((uint32_t)sc[aw.y & 0xffffu] << 5) | ((uint32_t)sc[aw.y >> 16] << 4) |
              ((uint32_t)sc[aw.z & 0xffffu] << 3) | ((uint32_t)sc[aw.z >> 16] << 2) |
              ((uint32_t)sc[aw.w & 0xffffu] << 1) |  (uint32_t)sc[aw.w >> 16];
          uint32_t w = lutbits[n * 8 + (idx >> 5)];
          sn[n] = (uint8_t)((w >> (idx & 31u)) & 1u);
        }
        __syncthreads();
        cur ^= 1;
      }
    }
  }

  // write reservoir states as f16, swizzled into WMMA 16-bit A layout:
  // k = 32*kb + koff ; koff -> half = (koff>>3)&1, e = (koff&7)|((koff>>4)<<3)
  const uint8_t* sc = sbuf[cur];
  for (int r = tid; r < R_N; r += 1024) {
    int kb = r >> 5, koff = r & 31;
    int h  = (koff >> 3) & 1;
    int e  = (koff & 7) | ((koff >> 4) << 3);
    uint16_t hv = sc[I_N + r] ? (uint16_t)0x3C00 : (uint16_t)0;  // f16 1.0 / 0.0
    sres_sw[((m * 512 + kb) * 2 + h) * 16 + e] = hv;
  }
}

// ------------------------------------------------------------------
// Readout: out = sigmoid(S(64x16384) @ W_out^T + b) via v_wmma_f32_16x16x32_f16.
// One block per 16-row M tile; 8 waves split K; LDS reduction.
// ------------------------------------------------------------------
__global__ __launch_bounds__(256)
void readout_kernel(const _Float16* __restrict__ sres_sw,
                    const _Float16* __restrict__ Wh,
                    const float* __restrict__ b_out,
                    float* __restrict__ out) {
  __shared__ float red[8][8][32];
  const int mtile = blockIdx.x;           // 0..3
  const int tid   = threadIdx.x;
  const int wave  = tid >> 5;             // 0..7
  const int lane  = tid & 31;
  const int row   = lane & 15;            // A row within tile == batch row
  const int half  = lane >> 4;
  const int col   = lane & 15;            // B column (output channel)
  const int mglob = mtile * 16 + row;

  v8f acc = {};
  const v16h* Ap = (const v16h*)sres_sw;  // groups of 16 halves (32B, aligned)
  for (int kb = wave; kb < 512; kb += 8) {
    v16h a = Ap[(mglob * 512 + kb) * 2 + half];                       // swizzled states
    v16h b;                                                           // K = 8*half + e (e<8), 16 + 8*half + (e-8) (e>=8)
    #pragma unroll
    for (int i = 0; i < 8; ++i) {
      b[i]     = Wh[col * R_N + kb * 32 + half * 8 + i];
      b[8 + i] = Wh[col * R_N + kb * 32 + 16 + half * 8 + i];
    }
    acc = __builtin_amdgcn_wmma_f32_16x16x32_f16(
        /*neg_a=*/false, a, /*neg_b=*/false, b,
        /*c_mod=*/(short)0, acc, /*reuse_a=*/false, /*reuse_b=*/false);
  }
  #pragma unroll
  for (int r = 0; r < 8; ++r) red[wave][r][lane] = acc[r];
  __syncthreads();

  // each of the 256 threads owns one (r,lane) element of the 16x16 D tile
  const int rr = tid >> 5;
  float sum = 0.f;
  #pragma unroll
  for (int w = 0; w < 8; ++w) sum += red[w][rr][lane];
  const int M = rr + ((lane >> 4) << 3);  // D layout: M = r + 8*(lane>=16)
  const int Nc = lane & 15;
  if (Nc < NOUT) {
    float v = sum + b_out[Nc];
    out[(mtile * 16 + M) * NOUT + Nc] = 1.f / (1.f + expf(-v));
  }
}

extern "C" void kernel_launch(void* const* d_in, const int* in_sizes, int n_in,
                              void* d_out, int out_size, void* d_ws, size_t ws_size,
                              hipStream_t stream) {
  (void)in_sizes; (void)n_in; (void)out_size; (void)ws_size;
  const float* x              = (const float*)d_in[0];
  const float* w_in           = (const float*)d_in[1];
  const int*   adj_list       = (const int*)d_in[2];
  /* d_in[3] adj_mask: derived from deg instead (bool dtype) */
  const int*   deg            = (const int*)d_in[4];
  const float* lut            = (const float*)d_in[5];
  /* d_in[6] powers_of_2: hardcoded */
  const float* initial_states = (const float*)d_in[7];
  const float* W_out          = (const float*)d_in[8];
  const float* b_out          = (const float*)d_in[9];

  uint8_t* ws = (uint8_t*)d_ws;
  uint32_t*  wbits   = (uint32_t*)(ws + OFF_WBITS);
  uint32_t*  xbits   = (uint32_t*)(ws + OFF_XBITS);
  uint16_t*  adj16   = (uint16_t*)(ws + OFF_ADJ16);
  uint32_t*  lutbits = (uint32_t*)(ws + OFF_LUTB);
  _Float16*  Wh      = (_Float16*)(ws + OFF_WH);
  uint16_t*  sres    = (uint16_t*)(ws + OFF_SRES);

  // total prep items: 1024 + 2048 + 135168 + 135168 + 262144 = 535552 = 2092*256
  prep_kernel<<<2092, 256, 0, stream>>>(x, w_in, adj_list, deg, lut, W_out,
                                        wbits, xbits, adj16, lutbits, Wh);
  reservoir_kernel<<<MB, 1024, 0, stream>>>(initial_states, wbits, xbits,
                                            adj16, lutbits, sres);
  readout_kernel<<<4, 256, 0, stream>>>((const _Float16*)sres, Wh, b_out,
                                        (float*)d_out);
}
